// GuidedMambaVSSBlock_32719060861029
// MI455X (gfx1250) — matrix operand order repaired and checked by reading, not compiled, not run on hardware
//
#include <hip/hip_runtime.h>
#include <hip/hip_bf16.h>
#include <math.h>

// ---------------------------------------------------------------------------
// GuidedMambaVSSBlock for MI455X (gfx1250, wave32, WMMA).
// All big GEMMs (in_proj 16384x384x96, x_proj 4x 16384x32(22)x192,
// out_proj 16384x96x192) use v_wmma_f32_16x16x32_f16.
// Weights are zero-padded to N%16==0 at f16-conversion time so the WMMA
// K-loop has no EXEC divergence (pure global_load_b128 + v_wmma).
// Selective scan: 16 blocks (b,k) x 192 threads (one Di-channel each),
// 8 recurrent states in registers, per-step (dt,B,C) broadcast via LDS.
// ---------------------------------------------------------------------------

typedef __attribute__((ext_vector_type(16))) _Float16 v16h;
typedef __attribute__((ext_vector_type(8)))  _Float16 v8h;
typedef __attribute__((ext_vector_type(8)))  float    v8f;

#define Bv   4
#define Cv   96
#define Hv   64
#define Wv   64
#define Lv   4096
#define Div  192
#define Nst  8
#define Rv   6
#define Kdir 4
#define DBLC 22   // R + 2n
#define DBLP 32   // DBLC padded to multiple of 16 for WMMA

__device__ __forceinline__ float wave_sum(float v) {
  for (int m = 16; m >= 1; m >>= 1) v += __shfl_xor(v, m, 32);
  return v;
}

// ---------------- stage 1: instance norm stats + masked pooled -------------
__global__ void inorm_pool_kernel(const float* __restrict__ x,
                                  const float* __restrict__ dz,
                                  float* __restrict__ mean, float* __restrict__ rstd,
                                  float* __restrict__ pooled) {
  int bc = blockIdx.x;            // b*96 + c
  int b  = bc / Cv;
  const float* xp  = x  + (size_t)bc * Lv;
  const float* dzp = dz + (size_t)b  * Lv;
  float s = 0.f, s2 = 0.f, sxd = 0.f, sd = 0.f;
  for (int i = threadIdx.x; i < Lv; i += 256) {
    float v = xp[i], dv = dzp[i];
    s += v; s2 += v * v; sxd += v * dv; sd += dv;
  }
  __shared__ float r0[256], r1[256], r2[256], r3[256];
  r0[threadIdx.x] = s; r1[threadIdx.x] = s2; r2[threadIdx.x] = sxd; r3[threadIdx.x] = sd;
  __syncthreads();
  for (int st = 128; st > 0; st >>= 1) {
    if (threadIdx.x < st) {
      r0[threadIdx.x] += r0[threadIdx.x + st];
      r1[threadIdx.x] += r1[threadIdx.x + st];
      r2[threadIdx.x] += r2[threadIdx.x + st];
      r3[threadIdx.x] += r3[threadIdx.x + st];
    }
    __syncthreads();
  }
  if (threadIdx.x == 0) {
    float m   = r0[0] / (float)Lv;
    float var = r1[0] / (float)Lv - m * m;
    float rs  = rsqrtf(var + 1e-5f);
    mean[bc] = m; rstd[bc] = rs;
    pooled[bc] = rs * (r2[0] - m * r3[0]) / (r3[0] + 1e-6f);
  }
}

// ---------------- stage 2: CCA squeeze-excite MLP --------------------------
__global__ void cca_kernel(const float* __restrict__ pooled,
                           const float* __restrict__ w1, const float* __restrict__ b1,
                           const float* __restrict__ w2, const float* __restrict__ b2,
                           float* __restrict__ sbuf) {
  int b = blockIdx.x, t = threadIdx.x;   // 96 threads
  __shared__ float p[Cv], hid[16];
  p[t] = pooled[b * Cv + t];
  __syncthreads();
  if (t < 16) {
    float a = b1[t];
    for (int c = 0; c < Cv; ++c) a += p[c] * w1[t * Cv + c];
    hid[t] = fmaxf(a, 0.f);
  }
  __syncthreads();
  float a = b2[t];
  for (int q = 0; q < 16; ++q) a += hid[q] * w2[t * 16 + q];
  sbuf[b * Cv + t] = 1.f / (1.f + expf(-a));
}

// -------- stage 3: x = xn*s + 0.4x ; NHWC ; per-pixel LayerNorm ------------
__global__ void fuse_ln_kernel(const float* __restrict__ x,
                               const float* __restrict__ mean, const float* __restrict__ rstd,
                               const float* __restrict__ sbuf,
                               const float* __restrict__ lng, const float* __restrict__ lnb,
                               float* __restrict__ xh, _Float16* __restrict__ xln) {
  int wave = threadIdx.x >> 5, lane = threadIdx.x & 31;
  int pix = blockIdx.x * 8 + wave;           // b*L + l
  int b = pix >> 12, l = pix & 4095;
  float v[3], s = 0.f, s2 = 0.f;
  for (int j = 0; j < 3; ++j) {
    int c = lane + 32 * j;
    float xv = x[((size_t)b * Cv + c) * Lv + l];
    float xn = (xv - mean[b * Cv + c]) * rstd[b * Cv + c];
    float x2 = xn * sbuf[b * Cv + c] + 0.4f * xv;
    v[j] = x2; s += x2; s2 += x2 * x2;
  }
  s = wave_sum(s); s2 = wave_sum(s2);
  float mu = s / (float)Cv;
  float rs = rsqrtf(s2 / (float)Cv - mu * mu + 1e-5f);
  for (int j = 0; j < 3; ++j) {
    int c = lane + 32 * j;
    xh[(size_t)pix * Cv + c] = v[j];
    xln[(size_t)pix * Cv + c] = (_Float16)((v[j] - mu) * rs * lng[c] + lnb[c]);
  }
}

// ---------------- f32 -> f16 weight conversion -----------------------------
__global__ void f2h_kernel(const float* __restrict__ src, _Float16* __restrict__ dst, int n) {
  int i = blockIdx.x * 256 + threadIdx.x;
  if (i < n) dst[i] = (_Float16)src[i];
}

// x_proj_w (4,22,192) -> (4,32,192) f16, rows 22..31 zero-padded
__global__ void f2h_pad_xproj_kernel(const float* __restrict__ src, _Float16* __restrict__ dst) {
  int i = blockIdx.x * 256 + threadIdx.x;          // over 4*32*192
  if (i >= Kdir * DBLP * Div) return;
  int k = i % Div, n = (i / Div) % DBLP, dir = i / (DBLP * Div);
  dst[i] = (n < DBLC) ? (_Float16)src[((size_t)dir * DBLC + n) * Div + k] : (_Float16)0.f;
}

// ---------------- WMMA GEMM: C[M,N] = A[M,K] * W[N,K]^T --------------------
// A: f16 row-major (M,K); W: f16 row-major (N,K), N % 16 == 0 (zero-padded);
// K % 32 == 0; M % 128 == 0. Block = 256 threads = 8 waves; each wave owns a
// 16x16 output tile. Inner loop is divergence-free (EXEC all-ones for WMMA).
// epi 0: out[m*Nvalid+n]
// epi 1: in_proj split -> out = xin NCHW (n<192), out2 = z (n>=192)
// epi 2: x_dbl layout out[((b*4+kdir)*L + l)*22 + n]
__global__ void gemm_wmma_kernel(const _Float16* __restrict__ A,
                                 const _Float16* __restrict__ W,
                                 float* __restrict__ out, float* __restrict__ out2,
                                 int M, int Nvalid, int K, int epi, int kdir) {
  const int lane = threadIdx.x & 31;
  const int wave = threadIdx.x >> 5;
  const int mBase = blockIdx.x * 128 + wave * 16;
  const int nBase = blockIdx.y * 16;

  const int hi = (lane >= 16);
  const int mA = mBase + (lane & 15);          // A row held by this lane
  const int aKoff = hi ? 8 : 0;                // ISA 16-bit A 16x32 layout
  const int nCol = nBase + (lane & 15);        // B column held by this lane
  const int bKoff = hi ? 16 : 0;               // contiguous k = 16h + s, used for A and B alike
  const _Float16* __restrict__ Arow = A + (size_t)mA * K;
  const _Float16* __restrict__ Wrow = W + (size_t)nCol * K;

  v8f acc = {};
  for (int k0 = 0; k0 < K; k0 += 32) {
    union { v16h v; v8h h[2]; } a;
    a.h[0] = *(const v8h*)(Arow + k0 + bKoff);
    a.h[1] = *(const v8h*)(Arow + k0 + bKoff + 8);
    v16h bfrag = *(const v16h*)(Wrow + k0 + bKoff);
    __builtin_prefetch((const void*)(Arow + k0 + 32), 0, 3);   // speculative
    acc = __builtin_amdgcn_wmma_f32_16x16x32_f16(false, a.v, false, bfrag,
                                                 (short)0, acc, false, false);
  }
  if (nCol >= Nvalid) return;
  for (int v = 0; v < 8; ++v) {
    int m = mBase + v + (hi ? 8 : 0);          // ISA 32-bit C/D layout
    float val = acc[v];
    if (epi == 0) {
      out[(size_t)m * Nvalid + nCol] = val;
    } else if (epi == 1) {
      int b = m >> 12, l = m & 4095;
      if (nCol < Div) out[((size_t)b * Div + nCol) * Lv + l] = val;       // xin NCHW
      else            out2[(size_t)m * Div + (nCol - Div)] = val;         // z
    } else {
      int b = m >> 12, l = m & 4095;
      out[((size_t)(b * Kdir + kdir) * Lv + l) * DBLC + nCol] = val;      // x_dbl
    }
  }
}

// ---------------- depthwise 3x3 conv + SiLU; emit row & transposed f16 -----
__global__ void dwconv_silu_kernel(const float* __restrict__ xin,
                                   const float* __restrict__ cw, const float* __restrict__ cb,
                                   _Float16* __restrict__ uh0, _Float16* __restrict__ uh1) {
  int pix = blockIdx.x;                 // b*L + l
  int b = pix >> 12, l = pix & 4095;
  int h = l >> 6, w = l & 63;
  int d = threadIdx.x;                  // 0..191
  const float* xp = xin + ((size_t)b * Div + d) * Lv;
  float acc = cb[d];
  #pragma unroll
  for (int dy = -1; dy <= 1; ++dy)
    #pragma unroll
    for (int dx = -1; dx <= 1; ++dx) {
      int hh = h + dy, ww = w + dx;
      if (hh >= 0 && hh < Hv && ww >= 0 && ww < Wv)
        acc += xp[hh * Wv + ww] * cw[d * 9 + (dy + 1) * 3 + (dx + 1)];
    }
  float v = acc / (1.f + expf(-acc));   // SiLU
  uh0[((size_t)b * Lv + (h * Wv + w)) * Div + d] = (_Float16)v;
  uh1[((size_t)b * Lv + (w * Hv + h)) * Div + d] = (_Float16)v;
}

// ---------------- selective scan: 16 blocks (b,k) x 192 threads ------------
__global__ void scan_kernel(const _Float16* __restrict__ uh0,
                            const _Float16* __restrict__ uh1,
                            const float* __restrict__ dbl,
                            const float* __restrict__ dt_w, const float* __restrict__ dt_b,
                            const float* __restrict__ A_logs, const float* __restrict__ Ds,
                            float* __restrict__ yk) {
  int b = blockIdx.x >> 2, k = blockIdx.x & 3;
  int d = threadIdx.x;                  // Di channel
  const _Float16* u = (k & 1) ? uh1 : uh0;
  const bool rev = (k >= 2);
  float wdt[Rv];
  #pragma unroll
  for (int r = 0; r < Rv; ++r) wdt[r] = dt_w[((size_t)k * Div + d) * Rv + r];
  float bias = dt_b[k * Div + d];
  float Ar[Nst];
  #pragma unroll
  for (int n = 0; n < Nst; ++n) Ar[n] = -expf(A_logs[((size_t)k * Div + d) * Nst + n]);
  float Dsv = Ds[k * Div + d];
  float hst[Nst];
  #pragma unroll
  for (int n = 0; n < Nst; ++n) hst[n] = 0.f;

  __shared__ float sh[DBLC];
  const float* dblb = dbl + (size_t)(b * Kdir + k) * Lv * DBLC;
  const size_t ubase = (size_t)b * Lv * Div;
  float* yout = yk + (size_t)k * ((size_t)Bv * Lv * Div) + ubase;

  for (int t = 0; t < Lv; ++t) {
    int li = rev ? (Lv - 1 - t) : t;    // position in scan-order arrays
    if (threadIdx.x < DBLC) sh[threadIdx.x] = dblb[(size_t)li * DBLC + threadIdx.x];
    __syncthreads();
    float uu = (float)u[ubase + (size_t)li * Div + d];
    float dt = bias;
    #pragma unroll
    for (int r = 0; r < Rv; ++r) dt += wdt[r] * sh[r];
    float delta = (dt > 20.f) ? dt : log1pf(expf(dt));   // softplus
    float du = delta * uu;
    float yacc = 0.f;
    #pragma unroll
    for (int n = 0; n < Nst; ++n) {
      float e = __expf(delta * Ar[n]);
      hst[n] = e * hst[n] + du * sh[Rv + n];
      yacc += hst[n] * sh[Rv + Nst + n];
    }
    int lo = (k & 1) ? ((li & 63) * 64 + (li >> 6)) : li;  // map back to row-major
    yout[(size_t)lo * Div + d] = yacc + Dsv * uu;
    __syncthreads();
  }
}

// --------- combine 4 directions + out-LN + SiLU(z) gate -> f16 -------------
__global__ void combine_gate_kernel(const float* __restrict__ yk,
                                    const float* __restrict__ z,
                                    const float* __restrict__ g, const float* __restrict__ bta,
                                    _Float16* __restrict__ gated) {
  const size_t S = (size_t)Bv * Lv * Div;
  int wave = threadIdx.x >> 5, lane = threadIdx.x & 31;
  int pix = blockIdx.x * 8 + wave;
  float vals[6], s = 0.f, s2 = 0.f;
  for (int j = 0; j < 6; ++j) {
    int dd = lane + 32 * j;
    size_t idx = (size_t)pix * Div + dd;
    float v = yk[idx] + yk[idx + S] + yk[idx + 2 * S] + yk[idx + 3 * S];
    vals[j] = v; s += v; s2 += v * v;
  }
  s = wave_sum(s); s2 = wave_sum(s2);
  float mu = s / (float)Div;
  float rs = rsqrtf(s2 / (float)Div - mu * mu + 1e-5f);
  for (int j = 0; j < 6; ++j) {
    int dd = lane + 32 * j;
    float y = (vals[j] - mu) * rs * g[dd] + bta[dd];
    float zv = z[(size_t)pix * Div + dd];
    gated[(size_t)pix * Div + dd] = (_Float16)(y * zv / (1.f + expf(-zv)));
  }
}

// --------- x3 = y_out + skip*xh, back to NCHW ------------------------------
__global__ void resid_kernel(const float* __restrict__ yout, const float* __restrict__ xh,
                             const float* __restrict__ skip, float* __restrict__ x3) {
  int bc = blockIdx.x;                 // b*96 + c
  int b = bc / Cv, c = bc % Cv;
  int l = blockIdx.y * 256 + threadIdx.x;
  size_t pi = ((size_t)b * Lv + l) * Cv + c;
  x3[((size_t)b * Cv + c) * Lv + l] = yout[pi] + skip[c] * xh[pi];
}

// --------- instance norm stats (no pool) -----------------------------------
__global__ void inorm2_kernel(const float* __restrict__ x,
                              float* __restrict__ mean, float* __restrict__ rstd) {
  int bc = blockIdx.x;
  const float* xp = x + (size_t)bc * Lv;
  float s = 0.f, s2 = 0.f;
  for (int i = threadIdx.x; i < Lv; i += 256) { float v = xp[i]; s += v; s2 += v * v; }
  __shared__ float r0[256], r1[256];
  r0[threadIdx.x] = s; r1[threadIdx.x] = s2;
  __syncthreads();
  for (int st = 128; st > 0; st >>= 1) {
    if (threadIdx.x < st) { r0[threadIdx.x] += r0[threadIdx.x + st]; r1[threadIdx.x] += r1[threadIdx.x + st]; }
    __syncthreads();
  }
  if (threadIdx.x == 0) {
    float m = r0[0] / (float)Lv;
    mean[bc] = m;
    rstd[bc] = rsqrtf(r1[0] / (float)Lv - m * m + 1e-5f);
  }
}

// --------- bottleneck: 1x1 (96->16) + relu ---------------------------------
__global__ void bnc1_kernel(const float* __restrict__ x3,
                            const float* __restrict__ mean, const float* __restrict__ rstd,
                            const float* __restrict__ w1, const float* __restrict__ b1,
                            float* __restrict__ h1) {
  int bc = blockIdx.x;                 // b*16 + cc
  int b = bc >> 4, cc = bc & 15;
  int l = blockIdx.y * 256 + threadIdx.x;
  float acc = b1[cc];
  for (int c = 0; c < Cv; ++c) {
    float xv = x3[((size_t)b * Cv + c) * Lv + l];
    acc += (xv - mean[b * Cv + c]) * rstd[b * Cv + c] * w1[cc * Cv + c];
  }
  h1[((size_t)b * 16 + cc) * Lv + l] = fmaxf(acc, 0.f);
}

// --------- bottleneck: 3x3 (16->16) + relu ---------------------------------
__global__ void bnc2_kernel(const float* __restrict__ h1,
                            const float* __restrict__ w2, const float* __restrict__ b2,
                            float* __restrict__ h2) {
  int bc = blockIdx.x;
  int b = bc >> 4, cc = bc & 15;
  int l = blockIdx.y * 256 + threadIdx.x;
  int h = l >> 6, w = l & 63;
  float acc = b2[cc];
  for (int ci = 0; ci < 16; ++ci) {
    const float* hp = h1 + ((size_t)b * 16 + ci) * Lv;
    #pragma unroll
    for (int dy = -1; dy <= 1; ++dy)
      #pragma unroll
      for (int dx = -1; dx <= 1; ++dx) {
        int hh = h + dy, ww = w + dx;
        if (hh >= 0 && hh < Hv && ww >= 0 && ww < Wv)
          acc += hp[hh * Wv + ww] * w2[((cc * 16 + ci) * 3 + (dy + 1)) * 3 + (dx + 1)];
      }
  }
  h2[((size_t)b * 16 + cc) * Lv + l] = fmaxf(acc, 0.f);
}

// --------- bottleneck: 1x1 (16->96) + skip ---------------------------------
__global__ void bnc3_kernel(const float* __restrict__ h2,
                            const float* __restrict__ w3, const float* __restrict__ b3,
                            const float* __restrict__ x3, float* __restrict__ out) {
  int bc = blockIdx.x;                 // b*96 + c
  int b = bc / Cv, c = bc % Cv;
  int l = blockIdx.y * 256 + threadIdx.x;
  float acc = b3[c];
  for (int ci = 0; ci < 16; ++ci)
    acc += h2[((size_t)b * 16 + ci) * Lv + l] * w3[c * 16 + ci];
  out[((size_t)b * Cv + c) * Lv + l] = acc + x3[((size_t)b * Cv + c) * Lv + l];
}

// ---------------------------------------------------------------------------
extern "C" void kernel_launch(void* const* d_in, const int* in_sizes, int n_in,
                              void* d_out, int out_size, void* d_ws, size_t ws_size,
                              hipStream_t stream) {
  (void)in_sizes; (void)n_in; (void)out_size; (void)ws_size;
  const float* x        = (const float*)d_in[0];
  const float* dz       = (const float*)d_in[1];
  const float* in_proj_w= (const float*)d_in[2];
  const float* conv_w   = (const float*)d_in[3];
  const float* conv_b   = (const float*)d_in[4];
  const float* x_proj_w = (const float*)d_in[5];
  const float* dt_w     = (const float*)d_in[6];
  const float* dt_b     = (const float*)d_in[7];
  const float* A_logs   = (const float*)d_in[8];
  const float* Ds       = (const float*)d_in[9];
  const float* out_ng   = (const float*)d_in[10];
  const float* out_nb   = (const float*)d_in[11];
  const float* out_proj_w=(const float*)d_in[12];
  const float* ln_g     = (const float*)d_in[13];
  const float* ln_b     = (const float*)d_in[14];
  const float* skip_ss  = (const float*)d_in[15];
  const float* cca_w1   = (const float*)d_in[16];
  const float* cca_b1   = (const float*)d_in[17];
  const float* cca_w2   = (const float*)d_in[18];
  const float* cca_b2   = (const float*)d_in[19];
  const float* bnc_w1   = (const float*)d_in[20];
  const float* bnc_b1   = (const float*)d_in[21];
  const float* bnc_w2   = (const float*)d_in[22];
  const float* bnc_b2   = (const float*)d_in[23];
  const float* bnc_w3   = (const float*)d_in[24];
  const float* bnc_b3   = (const float*)d_in[25];
  float* out = (float*)d_out;

  // ---- workspace carve-up (256B aligned regions) ----
  char* ws = (char*)d_ws;
  size_t off = 0;
  auto alloc = [&](size_t bytes) -> void* {
    void* p = ws + off;
    off = (off + bytes + 255) & ~(size_t)255;
    return p;
  };
  const size_t BLC = (size_t)Bv * Lv * Cv;     // 1,572,864
  const size_t BLD = (size_t)Bv * Lv * Div;    // 3,145,728
  float*    mean1  = (float*)alloc(Bv * Cv * 4);
  float*    rstd1  = (float*)alloc(Bv * Cv * 4);
  float*    pooled = (float*)alloc(Bv * Cv * 4);
  float*    sbuf   = (float*)alloc(Bv * Cv * 4);
  float*    mean2  = (float*)alloc(Bv * Cv * 4);
  float*    rstd2  = (float*)alloc(Bv * Cv * 4);
  _Float16* wh_in  = (_Float16*)alloc(2 * Div * Cv * 2);            // (384,96) f16
  _Float16* wh_xp  = (_Float16*)alloc(Kdir * DBLP * Div * 2);       // (4,32,192) padded
  _Float16* wh_out = (_Float16*)alloc(Cv * Div * 2);                // (96,192)
  float*    xh     = (float*)alloc(BLC * 4);                        // NHWC residual
  _Float16* xln    = (_Float16*)alloc(BLC * 2);                     // f16 GEMM input
  float*    xin    = (float*)alloc(BLD * 4);                        // NCHW conv input
  float*    z      = (float*)alloc(BLD * 4);                        // gate
  _Float16* uh0    = (_Float16*)alloc(BLD * 2);                     // row-major scan in
  _Float16* uh1    = (_Float16*)alloc(BLD * 2);                     // transposed scan in
  float*    dbl    = (float*)alloc((size_t)Bv * Kdir * Lv * DBLC * 4);
  float*    yk     = (float*)alloc(4 * BLD * 4);                    // per-direction ys
  float*    h1     = (float*)alloc((size_t)Bv * 16 * Lv * 4);
  float*    h2     = (float*)alloc((size_t)Bv * 16 * Lv * 4);
  // reuse dead regions:
  float*    x3     = xin;              // xin dead after dwconv (12.6MB region)
  _Float16* gated  = uh0;              // uh0 dead after scans
  float*    y_out  = (float*)uh1;      // uh1 dead after scans (same byte size)

  // 1) instance norm + masked pool
  inorm_pool_kernel<<<Bv * Cv, 256, 0, stream>>>(x, dz, mean1, rstd1, pooled);
  // 2) CCA MLP
  cca_kernel<<<Bv, 96, 0, stream>>>(pooled, cca_w1, cca_b1, cca_w2, cca_b2, sbuf);
  // 3) fuse + per-pixel LN -> xh (f32), xln (f16)
  fuse_ln_kernel<<<(Bv * Lv) / 8, 256, 0, stream>>>(x, mean1, rstd1, sbuf, ln_g, ln_b, xh, xln);
  // 4) weight conversion to f16 (x_proj zero-padded to N=32)
  f2h_kernel<<<(2 * Div * Cv + 255) / 256, 256, 0, stream>>>(in_proj_w, wh_in, 2 * Div * Cv);
  f2h_pad_xproj_kernel<<<(Kdir * DBLP * Div + 255) / 256, 256, 0, stream>>>(x_proj_w, wh_xp);
  f2h_kernel<<<(Cv * Div + 255) / 256, 256, 0, stream>>>(out_proj_w, wh_out, Cv * Div);
  // 5) in_proj WMMA GEMM: (16384,96) x (384,96)^T, split epilogue
  gemm_wmma_kernel<<<dim3(128, 24), 256, 0, stream>>>(xln, wh_in, xin, z,
                                                      Bv * Lv, 2 * Div, Cv, 1, 0);
  // 6) depthwise conv + SiLU -> uh0 (row-major), uh1 (HW-transposed)
  dwconv_silu_kernel<<<Bv * Lv, Div, 0, stream>>>(xin, conv_w, conv_b, uh0, uh1);
  // 7) x_proj WMMA GEMMs (one per direction; W padded to 32 rows)
  for (int k = 0; k < Kdir; ++k)
    gemm_wmma_kernel<<<dim3(128, 2), 256, 0, stream>>>((k & 1) ? uh1 : uh0,
                                                       wh_xp + (size_t)k * DBLP * Div,
                                                       dbl, nullptr,
                                                       Bv * Lv, DBLC, Div, 2, k);
  // 8) selective scan: 16 independent (b,k) scans
  scan_kernel<<<Bv * Kdir, Div, 0, stream>>>(uh0, uh1, dbl, dt_w, dt_b, A_logs, Ds, yk);
  // 9) combine + out-LN + SiLU gate -> gated f16
  combine_gate_kernel<<<(Bv * Lv) / 8, 256, 0, stream>>>(yk, z, out_ng, out_nb, gated);
  // 10) out_proj WMMA GEMM: (16384,192) x (96,192)^T
  gemm_wmma_kernel<<<dim3(128, 6), 256, 0, stream>>>(gated, wh_out, y_out, nullptr,
                                                     Bv * Lv, Cv, Div, 0, 0);
  // 11) residual -> NCHW
  resid_kernel<<<dim3(Bv * Cv, Lv / 256), 256, 0, stream>>>(y_out, xh, skip_ss, x3);
  // 12) instance norm stats
  inorm2_kernel<<<Bv * Cv, 256, 0, stream>>>(x3, mean2, rstd2);
  // 13-15) bottleneck convs + skip
  bnc1_kernel<<<dim3(Bv * 16, Lv / 256), 256, 0, stream>>>(x3, mean2, rstd2, bnc_w1, bnc_b1, h1);
  bnc2_kernel<<<dim3(Bv * 16, Lv / 256), 256, 0, stream>>>(h1, bnc_w2, bnc_b2, h2);
  bnc3_kernel<<<dim3(Bv * Cv, Lv / 256), 256, 0, stream>>>(h2, bnc_w3, bnc_b3, x3, out);
}
